// DKMIL_88304527605787
// MI455X (gfx1250) — compile-verified, matrix operand order checked by reading, not run on hardware
//
#include <hip/hip_runtime.h>
#include <hip/hip_bf16.h>
#include <math.h>

// ---------------------------------------------------------------------------
// DKMIL forward pass for MI455X (gfx1250, wave32, WMMA).
// All heavy GEMMs run through v_wmma_f32_16x16x32_bf16 (f32 accumulate),
// double-buffered LDS staging with register prefetch of the next K-tile.
// ---------------------------------------------------------------------------

typedef __bf16 bf16;
typedef __attribute__((ext_vector_type(4)))  __bf16 v4bf;
typedef __attribute__((ext_vector_type(8)))  __bf16 v8bf;
typedef __attribute__((ext_vector_type(16))) __bf16 v16bf;
typedef __attribute__((ext_vector_type(8)))  float  v8f;

union bfrag { v16bf v; v8bf h[2]; };

#define BN_SCALE 0.99999500003749969f   // (1+1e-5)^-0.5
#define NEG_BIG  (-3.402823466e38f)

enum { EPI_NONE = 0, EPI_LRELU, EPI_LRELU_BN, EPI_TANH, EPI_CDIST };

__device__ __forceinline__ float lrelu_f(float x) { return x > 0.f ? x : 0.01f * x; }

// ---------------------------------------------------------------------------
// WMMA GEMM: C[M,N] = epi(A x B + bias) (+resid). Tiles 128x128x32.
// TA: A stored [K,M] row-major (logical transpose).  TB: B stored [N,K].
// M,N multiples of 128; K multiple of 32 (true for every call site here).
// ---------------------------------------------------------------------------
#define BM 128
#define BN 128
#define BK 32
#define LDSD 40   // LDS row stride in bf16 (32 + 8 pad, keeps 16B alignment)

// Issue all global loads for one K-tile into registers (8x global_load_b128).
template <bool TA, bool TB>
__device__ __forceinline__ void stage_load(
    const float* __restrict__ A, const float* __restrict__ B,
    int lda, int ldb, int m0, int n0, int k0, int tid,
    float4 av[4], float4 bv[4])
{
#pragma unroll
    for (int it = 0; it < 4; ++it) {
        int fid = tid + it * 256;
        if (!TA) {
            int row = fid >> 3, kc = (fid & 7) << 2;
            av[it] = *(const float4*)(A + (size_t)(m0 + row) * lda + k0 + kc);
        } else {
            int kk = fid >> 5, mc = (fid & 31) << 2;
            av[it] = *(const float4*)(A + (size_t)(k0 + kk) * lda + m0 + mc);
        }
    }
#pragma unroll
    for (int it = 0; it < 4; ++it) {
        int fid = tid + it * 256;
        if (TB) {
            int row = fid >> 3, kc = (fid & 7) << 2;
            bv[it] = *(const float4*)(B + (size_t)(n0 + row) * ldb + k0 + kc);
        } else {
            int kk = fid >> 5, nc = (fid & 31) << 2;
            bv[it] = *(const float4*)(B + (size_t)(k0 + kk) * ldb + n0 + nc);
        }
    }
}

// Convert staged registers to bf16 and store into LDS tile buffers.
template <bool TA, bool TB>
__device__ __forceinline__ void stage_store(
    const float4 av[4], const float4 bv[4], int tid,
    bf16 (*As)[LDSD], bf16 (*Bs)[LDSD])
{
#pragma unroll
    for (int it = 0; it < 4; ++it) {
        int fid = tid + it * 256;
        v4bf p;
        p[0] = (bf16)av[it].x; p[1] = (bf16)av[it].y;
        p[2] = (bf16)av[it].z; p[3] = (bf16)av[it].w;
        if (!TA) {
            int row = fid >> 3, kc = (fid & 7) << 2;
            *(v4bf*)&As[row][kc] = p;                       // one 8B ds_store
        } else {
            int kk = fid >> 5, mc = (fid & 31) << 2;
            As[mc + 0][kk] = p[0]; As[mc + 1][kk] = p[1];
            As[mc + 2][kk] = p[2]; As[mc + 3][kk] = p[3];   // transpose scatter
        }
    }
#pragma unroll
    for (int it = 0; it < 4; ++it) {
        int fid = tid + it * 256;
        v4bf p;
        p[0] = (bf16)bv[it].x; p[1] = (bf16)bv[it].y;
        p[2] = (bf16)bv[it].z; p[3] = (bf16)bv[it].w;
        if (TB) {
            int row = fid >> 3, kc = (fid & 7) << 2;
            *(v4bf*)&Bs[row][kc] = p;
        } else {
            int kk = fid >> 5, nc = (fid & 31) << 2;
            Bs[nc + 0][kk] = p[0]; Bs[nc + 1][kk] = p[1];
            Bs[nc + 2][kk] = p[2]; Bs[nc + 3][kk] = p[3];
        }
    }
}

template <int EPI, bool TA, bool TB>
__global__ __launch_bounds__(256) void gemm_k(
    const float* __restrict__ A, const float* __restrict__ B, float* __restrict__ C,
    int M, int N, int K, int lda, int ldb, int ldc,
    const float* __restrict__ bias, const float* __restrict__ resid,
    const float* __restrict__ r1, const float* __restrict__ r2)
{
    __shared__ bf16 As[2][BM][LDSD];   // double buffered: 2 x 10 KB
    __shared__ bf16 Bs[2][BN][LDSD];

    const int tid  = threadIdx.x;
    const int lane = tid & 31;
    const int wid  = tid >> 5;
    const int wm   = wid & 3;     // 4 wave-rows of 32
    const int wn   = wid >> 2;    // 2 wave-cols of 64
    const int m0   = blockIdx.y * BM;
    const int n0   = blockIdx.x * BN;
    const int r    = lane & 15;
    const int half = lane >> 4;

    v8f acc[2][4];
#pragma unroll
    for (int i = 0; i < 2; i++)
#pragma unroll
        for (int j = 0; j < 4; j++)
#pragma unroll
            for (int e = 0; e < 8; e++) acc[i][j][e] = 0.f;

    float4 av[4], bv[4];
    stage_load<TA, TB>(A, B, lda, ldb, m0, n0, 0, tid, av, bv);
    stage_store<TA, TB>(av, bv, tid, As[0], Bs[0]);
    __syncthreads();

    int cur = 0;
    for (int k0 = 0; k0 < K; k0 += BK) {
        const bool has_next = (k0 + BK) < K;
        // prefetch next K-tile into registers while this tile computes
        if (has_next)
            stage_load<TA, TB>(A, B, lda, ldb, m0, n0, k0 + BK, tid, av, bv);

        // ---- fragments (ISA layout: VGPR0-3 <-> K=8h..8h+7, VGPR4-7 <-> K=16+8h..) ----
        bf16 (*Ac)[LDSD] = As[cur];
        bf16 (*Bc)[LDSD] = Bs[cur];
        bfrag a[2], b[4];
#pragma unroll
        for (int i = 0; i < 2; i++) {
            int row = wm * 32 + i * 16 + r;
            a[i].h[0] = *(const v8bf*)&Ac[row][8 * half];
            a[i].h[1] = *(const v8bf*)&Ac[row][16 + 8 * half];
        }
#pragma unroll
        for (int j = 0; j < 4; j++) {
            int col = wn * 64 + j * 16 + r;
            b[j].h[0] = *(const v8bf*)&Bc[col][8 * half];
            b[j].h[1] = *(const v8bf*)&Bc[col][16 + 8 * half];
        }
#pragma unroll
        for (int i = 0; i < 2; i++)
#pragma unroll
            for (int j = 0; j < 4; j++)
                acc[i][j] = __builtin_amdgcn_wmma_f32_16x16x32_bf16(
                    false, a[i].v, false, b[j].v, (short)0, acc[i][j], false, false);

        if (has_next)
            stage_store<TA, TB>(av, bv, tid, As[cur ^ 1], Bs[cur ^ 1]);
        __syncthreads();
        cur ^= 1;
    }

    // ---- fused epilogue + store (C/D layout: lane%16 = N, VGPR v = M offset) ----
#pragma unroll
    for (int i = 0; i < 2; i++)
#pragma unroll
        for (int j = 0; j < 4; j++) {
            int nn    = n0 + wn * 64 + j * 16 + r;
            int mbase = m0 + wm * 32 + i * 16 + half * 8;
#pragma unroll
            for (int v = 0; v < 8; ++v) {
                int   mm = mbase + v;
                float x  = acc[i][j][v];
                if (EPI == EPI_CDIST) {
                    x = r1[mm] + r2[nn] - 2.f * x;
                    x = sqrtf(fmaxf(x, 0.f));
                } else {
                    if (bias)  x += bias[nn];
                    if (resid) x += resid[(size_t)mm * ldc + nn];
                    if (EPI == EPI_LRELU || EPI == EPI_LRELU_BN) x = lrelu_f(x);
                    if (EPI == EPI_LRELU_BN) x *= BN_SCALE;
                    if (EPI == EPI_TANH) x = tanhf(x);
                }
                C[(size_t)mm * ldc + nn] = x;
            }
        }
}

static void gemm(hipStream_t s, const float* A, const float* B, float* C,
                 int M, int N, int K, int lda, int ldb, int ldc,
                 bool ta, bool tb, int epi,
                 const float* bias = nullptr, const float* resid = nullptr,
                 const float* r1 = nullptr, const float* r2 = nullptr)
{
    dim3 grid(N / BN, M / BM), blk(256);
#define GL(E, TAv, TBv) gemm_k<E, TAv, TBv><<<grid, blk, 0, s>>>(A, B, C, M, N, K, lda, ldb, ldc, bias, resid, r1, r2)
    if (ta) { GL(EPI_NONE, true, false); }
    else if (tb) {
        if (epi == EPI_CDIST) GL(EPI_CDIST, false, true);
        else                  GL(EPI_NONE,  false, true);
    } else {
        switch (epi) {
            case EPI_LRELU:    GL(EPI_LRELU,    false, false); break;
            case EPI_LRELU_BN: GL(EPI_LRELU_BN, false, false); break;
            case EPI_TANH:     GL(EPI_TANH,     false, false); break;
            default:           GL(EPI_NONE,     false, false); break;
        }
    }
#undef GL
}

// ---------------------------------------------------------------------------
// Column-wise softmax (over rows, per column) of X[M,N], in place.
// Block = 256 threads = 64 columns x 4 row-slabs.
// ---------------------------------------------------------------------------
__global__ void colsoftmax_k(float* X, int M, int N)
{
    __shared__ float red[4][64];
    int tid = threadIdx.x;
    int cl  = tid & 63;
    int ty  = tid >> 6;
    int c   = blockIdx.x * 64 + cl;

    float mx = NEG_BIG;
    for (int m = ty; m < M; m += 4) mx = fmaxf(mx, X[(size_t)m * N + c]);
    red[ty][cl] = mx; __syncthreads();
    if (ty == 0) {
        float a = fmaxf(fmaxf(red[0][cl], red[1][cl]), fmaxf(red[2][cl], red[3][cl]));
        red[0][cl] = a;
    }
    __syncthreads();
    mx = red[0][cl]; __syncthreads();

    float s = 0.f;
    for (int m = ty; m < M; m += 4) s += expf(X[(size_t)m * N + c] - mx);
    red[ty][cl] = s; __syncthreads();
    if (ty == 0) red[0][cl] = red[0][cl] + red[1][cl] + red[2][cl] + red[3][cl];
    __syncthreads();
    float inv = 1.f / red[0][cl];

    for (int m = ty; m < M; m += 4) {
        size_t o = (size_t)m * N + c;
        X[o] = expf(X[o] - mx) * inv;
    }
}

// out[m] = sum_j X[m,j]^2
__global__ void rownorm2_k(const float* __restrict__ X, float* __restrict__ out, int N)
{
    __shared__ float red[256];
    int m = blockIdx.x, t = threadIdx.x;
    float s = 0.f;
    for (int j = t; j < N; j += 256) { float v = X[(size_t)m * N + j]; s += v * v; }
    red[t] = s; __syncthreads();
    for (int off = 128; off > 0; off >>= 1) { if (t < off) red[t] += red[t + off]; __syncthreads(); }
    if (t == 0) out[m] = red[0];
}

// out[j] = sum_h A[h*N+j]*w[h] + b[0]    (A is [H,N])
__global__ void colvec_k(const float* __restrict__ A, const float* __restrict__ w,
                         const float* __restrict__ b, float* __restrict__ out, int H, int N)
{
    int j = blockIdx.x * 256 + threadIdx.x;
    if (j >= N) return;
    float s = b[0];
    for (int h = 0; h < H; h++) s += A[(size_t)h * N + j] * w[h];
    out[j] = s;
}

// iterative top-k (desc, ties -> lowest index), K<=4096, single block of 256
__global__ void topk_k(const float* __restrict__ sc, int K, int ntop, int* __restrict__ idx_out)
{
    __shared__ float sv[4096];
    __shared__ float rv[256];
    __shared__ int   ri[256];
    int t = threadIdx.x;
    for (int i = t; i < K; i += 256) sv[i] = sc[i];
    __syncthreads();
    for (int sel = 0; sel < ntop; ++sel) {
        float bv = NEG_BIG; int bi = K;
        for (int i = t; i < K; i += 256) {
            float v = sv[i];
            if (v > bv || (v == bv && i < bi)) { bv = v; bi = i; }
        }
        rv[t] = bv; ri[t] = bi; __syncthreads();
        for (int off = 128; off > 0; off >>= 1) {
            if (t < off) {
                if (rv[t + off] > rv[t] || (rv[t + off] == rv[t] && ri[t + off] < ri[t])) {
                    rv[t] = rv[t + off]; ri[t] = ri[t + off];
                }
            }
            __syncthreads();
        }
        if (t == 0) { idx_out[sel] = ri[0]; sv[ri[0]] = NEG_BIG; }
        __syncthreads();
    }
}

// per-bag first-argmax within [bounds[b], bounds[b+1])
__global__ void seg_argmax_k(const float* __restrict__ sc, const int* __restrict__ bounds,
                             int nbag, int* __restrict__ out)
{
    int b = blockIdx.x * blockDim.x + threadIdx.x;
    if (b >= nbag) return;
    int s = bounds[b], e = bounds[b + 1];
    float bv = NEG_BIG; int bi = s;
    for (int i = s; i < e; i++) { float v = sc[i]; if (v > bv) { bv = v; bi = i; } }
    out[b] = bi;
}

// out[m, 0..NG) = softmax_g( S[m, idx[g]] ); block = NG threads, one row/block
__global__ void gather_softmax_k(const float* __restrict__ S, int ldS,
                                 const int* __restrict__ idx, float* __restrict__ out, int NG)
{
    __shared__ float red[128];
    int m = blockIdx.x, t = threadIdx.x;
    float v = S[(size_t)m * ldS + idx[t]];
    red[t] = v; __syncthreads();
    for (int off = 64; off > 0; off >>= 1) { if (t < off) red[t] = fmaxf(red[t], red[t + off]); __syncthreads(); }
    float mx = red[0]; __syncthreads();
    float e = expf(v - mx);
    red[t] = e; __syncthreads();
    for (int off = 64; off > 0; off >>= 1) { if (t < off) red[t] += red[t + off]; __syncthreads(); }
    out[(size_t)m * NG + t] = e / red[0];
}

// X_cat[m, 0..768) = [ B[m,0..512) | bi[m,0..128) | bb[m,0..128) ]
__global__ void concat3_k(const float* __restrict__ B, const float* __restrict__ bi,
                          const float* __restrict__ bb, float* __restrict__ out, int M)
{
    int i = blockIdx.x * blockDim.x + threadIdx.x;
    int total = M * 768;
    if (i >= total) return;
    int m = i / 768, c = i % 768;
    float v;
    if (c < 512)      v = B [(size_t)m * 512 + c];
    else if (c < 640) v = bi[(size_t)m * 128 + (c - 512)];
    else              v = bb[(size_t)m * 128 + (c - 640)];
    out[i] = v;
}

// X[m, 0..1024) = [ Bf[m,0..512) | att[0..512) (broadcast) ]
__global__ void concat_tile_k(const float* __restrict__ Bf, const float* __restrict__ att,
                              float* __restrict__ out, int M)
{
    int i = blockIdx.x * blockDim.x + threadIdx.x;
    int total = M * 1024;
    if (i >= total) return;
    int m = i / 1024, c = i % 1024;
    out[i] = (c < 512) ? Bf[(size_t)m * 512 + c] : att[c - 512];
}

// out[m] = lrelu( sum_j V[m,j]*U[m,j]*w[j] + b[0] ), block = DA (=128) threads
__global__ void rowdot3_k(const float* __restrict__ V, const float* __restrict__ U,
                          const float* __restrict__ w, const float* __restrict__ b,
                          float* __restrict__ out, int DAv)
{
    __shared__ float red[128];
    int m = blockIdx.x, t = threadIdx.x;
    float s = V[(size_t)m * DAv + t] * U[(size_t)m * DAv + t] * w[t];
    red[t] = s; __syncthreads();
    for (int off = 64; off > 0; off >>= 1) { if (t < off) red[t] += red[t + off]; __syncthreads(); }
    if (t == 0) { float x = red[0] + b[0]; out[m] = lrelu_f(x); }
}

// out = softmax(in), length L, single block of 256
__global__ void softmax_vec_k(const float* __restrict__ in, float* __restrict__ out, int L)
{
    __shared__ float red[256];
    int t = threadIdx.x;
    float mx = NEG_BIG;
    for (int i = t; i < L; i += 256) mx = fmaxf(mx, in[i]);
    red[t] = mx; __syncthreads();
    for (int off = 128; off > 0; off >>= 1) { if (t < off) red[t] = fmaxf(red[t], red[t + off]); __syncthreads(); }
    mx = red[0]; __syncthreads();
    float s = 0.f;
    for (int i = t; i < L; i += 256) s += expf(in[i] - mx);
    red[t] = s; __syncthreads();
    for (int off = 128; off > 0; off >>= 1) { if (t < off) red[t] += red[t + off]; __syncthreads(); }
    float inv = 1.f / red[0];
    for (int i = t; i < L; i += 256) out[i] = expf(in[i] - mx) * inv;
}

// out[h] = sum_n a[n]*H[n,h]
__global__ void attn_pool_k(const float* __restrict__ a, const float* __restrict__ H,
                            float* __restrict__ out, int M, int Hd)
{
    int h = blockIdx.x * blockDim.x + threadIdx.x;
    if (h >= Hd) return;
    float s = 0.f;
    for (int n = 0; n < M; n++) s += a[n] * H[(size_t)n * Hd + h];
    out[h] = s;
}

// out[j] = [lrelu]( sum_i x[i]*W[i*Dout+j] + b[j] )
__global__ void vecmat_k(const float* __restrict__ x, const float* __restrict__ W,
                         const float* __restrict__ b, float* __restrict__ out,
                         int Din, int Dout, int do_lrelu)
{
    int j = blockIdx.x * blockDim.x + threadIdx.x;
    if (j >= Dout) return;
    float s = b[j];
    for (int i = 0; i < Din; i++) s += x[i] * W[(size_t)i * Dout + j];
    out[j] = do_lrelu ? lrelu_f(s) : s;
}

// Y_prob = clip(sigmoid(h . W + b)), Y_hat = (p>=0.5); h length 256
__global__ void final_k(const float* __restrict__ h, const float* __restrict__ W,
                        const float* __restrict__ b, float* __restrict__ out)
{
    __shared__ float red[256];
    int t = threadIdx.x;
    red[t] = h[t] * W[t]; __syncthreads();
    for (int off = 128; off > 0; off >>= 1) { if (t < off) red[t] += red[t + off]; __syncthreads(); }
    if (t == 0) {
        float p = 1.f / (1.f + expf(-(red[0] + b[0])));
        p = fminf(fmaxf(p, 1e-5f), 1.f - 1e-5f);
        out[0] = p;
        out[1] = (p >= 0.5f) ? 1.f : 0.f;
    }
}

// ---------------------------------------------------------------------------
// Host orchestration
// ---------------------------------------------------------------------------
extern "C" void kernel_launch(void* const* d_in, const int* in_sizes, int n_in,
                              void* d_out, int out_size, void* d_ws, size_t ws_size,
                              hipStream_t stream)
{
    (void)in_sizes; (void)n_in; (void)out_size; (void)ws_size;

    const int N = 4096, D = 512;
    const float* Bx   = (const float*)d_in[0];  // [1,4096,512]
    const float* KIns = (const float*)d_in[1];  // [1,4096,512]
    const float* BSp  = (const float*)d_in[2];  // [1,4096,512]
    const int*   iidx = (const int*)  d_in[3];  // [129]
    auto P = [&](int i) { return (const float*)d_in[i]; };
    // params (depth-first insertion order):
    // 4..9 ins_skip1, 10..15 ins_skip2, 16..21 ins_skip3, 22..29 ins_mask,
    // 30..35 bag_skip1, 36..41 bag_skip2, 42..47 bag_skip3, 48..55 bag_mask,
    // 56..57 fuse_emb, 58..67 att, 68..77 satt_att, 78..79 satt_emb, 80..81 clf

    // ---- workspace layout (floats) ----
    float* w = (float*)d_ws;
    size_t off = 0;
    auto alloc = [&](size_t n) { float* q = w + off; off += n; return q; };
    float* AFF  = alloc((size_t)N * N);        // affinity / residual
    float* FEAT = alloc((size_t)N * N);        // feat (softmaxed attention matrix)
    float* YB   = alloc((size_t)N * N);        // big Y3 for dd=4096 skip blocks
    float* S3   = alloc((size_t)N * N);        // skip3 output
    float* T1   = alloc((size_t)N * D);        // skip1 out
    float* T2   = alloc((size_t)N * D);        // skip2 out
    float* T3   = alloc((size_t)N * D);        // Y3 for dd=512 skip blocks
    float* Y1S  = alloc((size_t)N * 256);
    float* Y2S  = alloc((size_t)N * 256);
    float* R1v  = alloc(N);
    float* R2v  = alloc(N);
    float* MMb  = alloc(256 * 256);
    float* A3b  = alloc((size_t)256 * N);
    float* SCO  = alloc(N);                    // mask scores / attention weights scratch
    float* AVV  = alloc(N);                    // raw attention logits
    int*   IDXI = (int*)alloc(N);
    int*   SEGI = (int*)alloc(256);
    float* BINS = alloc((size_t)N * 128);      // B_ins (persists)
    float* BBAG = alloc((size_t)N * 128);      // B_bag (persists)
    float* XC1  = alloc((size_t)N * 768);
    float* BFUS = alloc((size_t)N * 512);      // B_fuse (persists)
    float* H1   = alloc((size_t)N * 256);
    float* VB   = alloc((size_t)N * 128);
    float* UB   = alloc((size_t)N * 128);
    float* MV   = alloc(1024);
    float* BATT = alloc(1024);                 // B_att [512]
    float* XC2  = alloc((size_t)N * 1024);
    float* H2   = alloc((size_t)N * 256);
    float* BV   = alloc(1024);
    float* BSA  = alloc(256);

    float* A_out = (float*)d_out + 2;          // satt attention weights [4096]

    // ---- skip_block: out = softmax_col(Y1@Y2^T) @ Y3 + X ----
    auto skip = [&](const float* X, int dd, int pb, float* out, float* y3) {
        gemm(stream, X, P(pb + 0), Y1S, N, 256, dd, dd, 256, 256, false, false, EPI_LRELU_BN, P(pb + 1));
        gemm(stream, X, P(pb + 2), Y2S, N, 256, dd, dd, 256, 256, false, false, EPI_LRELU_BN, P(pb + 3));
        gemm(stream, X, P(pb + 4), y3,  N, dd,  dd, dd, dd,  dd,  false, false, EPI_LRELU_BN, P(pb + 5));
        gemm(stream, Y1S, Y2S, FEAT, N, N, 256, 256, 256, N, false, true, EPI_NONE);
        colsoftmax_k<<<N / 64, 256, 0, stream>>>(FEAT, N, N);
        gemm(stream, FEAT, y3, out, N, dd, N, N, dd, dd, false, false, EPI_NONE, nullptr, X);
    };

    // ---- mask_block: scores (pre-softmax; rank-equivalent) -> SCO ----
    auto maskb = [&](const float* X, int dd, int pb) {
        gemm(stream, X, P(pb + 0), Y1S, N, 256, dd, dd, 256, 256, false, false, EPI_LRELU, P(pb + 1));
        gemm(stream, X, P(pb + 2), Y2S, N, 256, dd, dd, 256, 256, false, false, EPI_TANH,  P(pb + 3));
        gemm(stream, Y1S, Y2S, MMb, 256, 256, N, 256, 256, 256, true, false, EPI_NONE);          // X1^T @ X2
        gemm(stream, MMb, P(pb + 4), A3b, 256, dd, 256, 256, dd, dd, false, false, EPI_LRELU, P(pb + 5));
        colvec_k<<<dd / 256, 256, 0, stream>>>(A3b, P(pb + 6), P(pb + 7), SCO, 256, dd);
    };

    auto cdistf = [&](const float* Xs, const float* Ys) {
        rownorm2_k<<<N, 256, 0, stream>>>(Xs, R1v, D);
        rownorm2_k<<<N, 256, 0, stream>>>(Ys, R2v, D);
        gemm(stream, Xs, Ys, AFF, N, N, D, D, D, N, false, true, EPI_CDIST, nullptr, nullptr, R1v, R2v);
    };

    // ================= ins branch =================
    skip(Bx,   D, 4,  T1, T3);
    skip(KIns, D, 10, T2, T3);
    cdistf(T1, T2);
    skip(AFF, N, 16, S3, YB);
    maskb(S3, N, 22);
    topk_k<<<1, 256, 0, stream>>>(SCO, N, 128, IDXI);
    gather_softmax_k<<<N, 128, 0, stream>>>(S3, N, IDXI, BINS, 128);

    // ================= bag branch =================
    skip(Bx,  D, 30, T1, T3);
    skip(BSp, D, 36, T2, T3);
    cdistf(T1, T2);
    skip(AFF, N, 42, S3, YB);
    maskb(S3, N, 48);
    seg_argmax_k<<<1, 128, 0, stream>>>(SCO, iidx, 128, SEGI);
    gather_softmax_k<<<N, 128, 0, stream>>>(S3, N, SEGI, BBAG, 128);

    // ================= fuse =================
    concat3_k<<<(N * 768 + 255) / 256, 256, 0, stream>>>(Bx, BINS, BBAG, XC1, N);
    gemm(stream, XC1, P(56), BFUS, N, 512, 768, 768, 512, 512, false, false, EPI_LRELU, P(57));

    // ================= attention (att) =================
    gemm(stream, BFUS, P(58), H1, N, 256, 512, 512, 256, 256, false, false, EPI_LRELU, P(59));
    gemm(stream, H1,   P(60), VB, N, 128, 256, 256, 128, 128, false, false, EPI_TANH,  P(61));
    gemm(stream, H1,   P(62), UB, N, 128, 256, 256, 128, 128, false, false, EPI_LRELU, P(63));
    rowdot3_k<<<N, 128, 0, stream>>>(VB, UB, P(64), P(65), AVV, 128);
    softmax_vec_k<<<1, 256, 0, stream>>>(AVV, SCO, N);
    attn_pool_k<<<1, 256, 0, stream>>>(SCO, H1, MV, N, 256);
    vecmat_k<<<2, 256, 0, stream>>>(MV, P(66), P(67), BATT, 256, 512, 1);

    // ================= self-attention (satt) =================
    concat_tile_k<<<(N * 1024 + 255) / 256, 256, 0, stream>>>(BFUS, BATT, XC2, N);
    gemm(stream, XC2, P(68), H2, N, 256, 1024, 1024, 256, 256, false, false, EPI_LRELU, P(69));
    gemm(stream, H2,  P(70), VB, N, 128, 256,  256,  128, 128, false, false, EPI_TANH,  P(71));
    gemm(stream, H2,  P(72), UB, N, 128, 256,  256,  128, 128, false, false, EPI_LRELU, P(73));
    rowdot3_k<<<N, 128, 0, stream>>>(VB, UB, P(74), P(75), AVV, 128);
    softmax_vec_k<<<1, 256, 0, stream>>>(AVV, A_out, N);          // output A
    attn_pool_k<<<1, 256, 0, stream>>>(A_out, H2, MV, N, 256);
    vecmat_k<<<4, 256, 0, stream>>>(MV, P(76), P(77), BV, 256, 1024, 1);   // b = lrelu(Mv@We2+be2)
    vecmat_k<<<1, 256, 0, stream>>>(BV, P(78), P(79), BSA, 1024, 256, 1);  // satt_emb
    final_k<<<1, 256, 0, stream>>>(BSA, P(80), P(81), (float*)d_out);      // Y_prob, Y_hat
}
